// ROIHead_35399120454289
// MI455X (gfx1250) — compile-verified, matrix operand order checked by reading, not run on hardware
//
#include <hip/hip_runtime.h>
#include <hip/hip_bf16.h>
#include <math.h>

// ---------------------------------------------------------------------------
// ROI head for MI455X (gfx1250, wave32, WMMA + async global->LDS).
// 5x GEMM 8192x2048x2048 in bf16 WMMA (f32 accum), double-buffered LDS fed by
// GLOBAL_LOAD_ASYNC_TO_LDS_B128 (ASYNCcnt), 128x128 block tile, 8 waves each
// owning 32x64 (8 WMMAs/K-step). Weights L2-resident (8MB bf16, L2=192MB).
// ---------------------------------------------------------------------------

#define N_ROIS      8192
#define IN_CH       2048
#define NUM_CLASSES 81
#define N_GT        64
#define IOU_THR     0.7f
#define BG_THR      0.8f

typedef __bf16 bf16_t;
typedef __attribute__((ext_vector_type(8)))  bf16_t v8bf;
typedef __attribute__((ext_vector_type(16))) bf16_t v16bf;
typedef __attribute__((ext_vector_type(8)))  float  v8f;

// ---- CDNA5 async copy: 16B global -> LDS, tracked by ASYNCcnt ----
__device__ __forceinline__ void async_copy16(unsigned lds_off, const void* gptr) {
  asm volatile("global_load_async_to_lds_b128 %0, %1, off"
               :: "v"(lds_off), "v"(gptr) : "memory");
}
__device__ __forceinline__ void wait_async0() {
  asm volatile("s_wait_asynccnt 0x0" ::: "memory");
}

// ---------------- f32 -> bf16 conversion (grid-stride) ----------------
__global__ void k_f32_to_bf16(const float* __restrict__ in, bf16_t* __restrict__ out, int n) {
  int i = blockIdx.x * blockDim.x + threadIdx.x;
  int stride = gridDim.x * blockDim.x;
  for (; i < n; i += stride) out[i] = (bf16_t)in[i];
}

// ---------------- bf16 WMMA GEMM: C[M,N] = A[M,K] @ W[N,K]^T + bias ----------------
// Block tile 128(M) x 128(N), K-step 32, double-buffered LDS (async fill).
// 8 waves in 4(M) x 2(N); each wave owns 32x64 = 2x4 WMMA 16x16 tiles.
#define BM 128
#define BN 128
#define BK 32
#define LDT 40   // padded LDS row stride (bf16 elems); 80B, multiple of 16B

__global__ __launch_bounds__(256) void k_gemm_bf16(
    const bf16_t* __restrict__ A,   // M x K row-major
    const bf16_t* __restrict__ W,   // N x K row-major (B = W^T, K-contiguous)
    const float*  __restrict__ bias,
    float*  __restrict__ Cf,        // optional f32 output (nullptr to skip)
    bf16_t* __restrict__ Cb,        // optional bf16 output (nullptr to skip)
    int M, int N, int K)
{
  __shared__ __align__(16) bf16_t As[2][BM * LDT];
  __shared__ __align__(16) bf16_t Bs[2][BN * LDT];

  const int tid  = threadIdx.x;
  const int lane = tid & 31;
  const int wave = tid >> 5;        // 0..7
  const int wm   = wave >> 1;       // 0..3 : 32-row band
  const int wn   = wave & 1;        // 0..1 : 64-col band
  const int bm   = blockIdx.y * BM;
  const int bn   = blockIdx.x * BN;

  const int col  = lane & 15;       // fragment row / output N
  const int hsel = lane >> 4;       // K-half select per ISA 16-bit layout
  const int kb0  = 8 * hsel;        // half0 -> K 0..7 & 16..23 ; half1 -> 8..15 & 24..31

  // tile-copy coordinates: 16B chunks; 1024 chunks per (A+B) tile, 4 per thread
  const int cRow = tid >> 2;        // 0..63
  const int cCol = (tid & 3) * 8;   // 0,8,16,24

  const bf16_t* gA0 = &A[(size_t)(bm + cRow)      * K + cCol];
  const bf16_t* gA1 = &A[(size_t)(bm + cRow + 64) * K + cCol];
  const bf16_t* gB0 = &W[(size_t)(bn + cRow)      * K + cCol];
  const bf16_t* gB1 = &W[(size_t)(bn + cRow + 64) * K + cCol];

  auto issue_tile = [&](int k0, int buf) {
    async_copy16((unsigned)(uintptr_t)&As[buf][(cRow)      * LDT + cCol], gA0 + k0);
    async_copy16((unsigned)(uintptr_t)&As[buf][(cRow + 64) * LDT + cCol], gA1 + k0);
    async_copy16((unsigned)(uintptr_t)&Bs[buf][(cRow)      * LDT + cCol], gB0 + k0);
    async_copy16((unsigned)(uintptr_t)&Bs[buf][(cRow + 64) * LDT + cCol], gB1 + k0);
  };

  // prologue: fill buffer 0
  issue_tile(0, 0);
  wait_async0();
  __syncthreads();

  v8f acc[2][4] = {};
  int buf = 0;
  for (int k0 = 0; k0 < K; k0 += BK, buf ^= 1) {
    if (k0 + BK < K) issue_tile(k0 + BK, buf ^ 1);   // overlap next-tile DMA with math

    // ---- fragments from LDS per ISA 16-bit A/B layout ----
    v16bf afr[2], bfr[4];
#pragma unroll
    for (int mi = 0; mi < 2; ++mi) {
      int r = wm * 32 + mi * 16 + col;
      v8bf lo = *(const v8bf*)&As[buf][r * LDT + kb0];
      v8bf hi = *(const v8bf*)&As[buf][r * LDT + kb0 + 16];
      afr[mi] = __builtin_shufflevector(lo, hi, 0,1,2,3,4,5,6,7,8,9,10,11,12,13,14,15);
    }
#pragma unroll
    for (int ni = 0; ni < 4; ++ni) {
      int r = wn * 64 + ni * 16 + col;
      v8bf lo = *(const v8bf*)&Bs[buf][r * LDT + kb0];
      v8bf hi = *(const v8bf*)&Bs[buf][r * LDT + kb0 + 16];
      bfr[ni] = __builtin_shufflevector(lo, hi, 0,1,2,3,4,5,6,7,8,9,10,11,12,13,14,15);
    }

#pragma unroll
    for (int mi = 0; mi < 2; ++mi)
#pragma unroll
      for (int ni = 0; ni < 4; ++ni)
        acc[mi][ni] = __builtin_amdgcn_wmma_f32_16x16x32_bf16(
            false, afr[mi], false, bfr[ni], (short)0, acc[mi][ni], false, false);

    wait_async0();        // our async writes to buf^1 have landed in LDS
    __syncthreads();      // publish to all waves; also fences this buf's reads
  }

  // ---- epilogue: D layout lane->(N=lane&15, Mhalf=lane>>4), VGPR r -> M=r+8*half ----
  if (Cb) {
#pragma unroll
    for (int mi = 0; mi < 2; ++mi)
#pragma unroll
      for (int ni = 0; ni < 4; ++ni) {
        int gn = bn + wn * 64 + ni * 16 + col;
        float bv = bias[gn];
#pragma unroll
        for (int r = 0; r < 8; ++r) {
          int gm = bm + wm * 32 + mi * 16 + r + 8 * hsel;
          Cb[(size_t)gm * N + gn] = (bf16_t)(acc[mi][ni][r] + bv);
        }
      }
  }
  if (Cf) {
#pragma unroll
    for (int mi = 0; mi < 2; ++mi)
#pragma unroll
      for (int ni = 0; ni < 4; ++ni) {
        int gn = bn + wn * 64 + ni * 16 + col;
        float bv = bias[gn];
#pragma unroll
        for (int r = 0; r < 8; ++r) {
          int gm = bm + wm * 32 + mi * 16 + r + 8 * hsel;
          Cf[(size_t)gm * N + gn] = acc[mi][ni][r] + bv;
        }
      }
  }
}

// ---------------- small head GEMM: C[M,N] = A_bf16[M,K] @ W_f32[N,K]^T + b ----------------
__global__ __launch_bounds__(128) void k_head_gemm(
    const bf16_t* __restrict__ A, const float* __restrict__ W,
    const float* __restrict__ bias, float* __restrict__ C, int N, int K)
{
  __shared__ float a_sh[IN_CH];
  const int row = blockIdx.x;
  for (int k = threadIdx.x; k < K; k += 128) a_sh[k] = (float)A[(size_t)row * K + k];
  __syncthreads();
  for (int c = threadIdx.x; c < N; c += 128) {
    const float* w = W + (size_t)c * K;
    float acc = 0.f;
    for (int k = 0; k < K; ++k) acc = fmaf(a_sh[k], w[k], acc);
    C[(size_t)row * N + c] = acc + bias[c];
  }
}

// ---------------- softmax over 81 classes; score = max prob = 1/sum(exp(x-max)) ----------------
__global__ __launch_bounds__(128) void k_softmax(const float* __restrict__ logits,
                                                 float* __restrict__ probs,
                                                 float* __restrict__ scores)
{
  __shared__ float sh[NUM_CLASSES];
  __shared__ float smax, ssum;
  const int row = blockIdx.x, t = threadIdx.x;
  if (t < NUM_CLASSES) sh[t] = logits[(size_t)row * NUM_CLASSES + t];
  __syncthreads();
  if (t == 0) { float m = sh[0]; for (int i = 1; i < NUM_CLASSES; ++i) m = fmaxf(m, sh[i]); smax = m; }
  __syncthreads();
  if (t < NUM_CLASSES) sh[t] = __expf(sh[t] - smax);
  __syncthreads();
  if (t == 0) { float s = 0.f; for (int i = 0; i < NUM_CLASSES; ++i) s += sh[i]; ssum = s; scores[row] = 1.0f / s; }
  __syncthreads();
  if (t < NUM_CLASSES) probs[(size_t)row * NUM_CLASSES + t] = sh[t] / ssum;
}

// ---------------- loc2box decode + nms_boxes (xyxy) ----------------
__global__ void k_pred(const float* __restrict__ rois, const float* __restrict__ loc,
                       float* __restrict__ pred_box, float* __restrict__ nms_boxes)
{
  int i = blockIdx.x * blockDim.x + threadIdx.x;
  if (i >= N_ROIS) return;
  float r0 = rois[i*4+0], r1 = rois[i*4+1], r2 = rois[i*4+2], r3 = rois[i*4+3];
  float h = r3 - r1, w = r2 - r0;                       // rois_ywh = (y=r1, x=r0, h, w)
  float cy = r1 + 0.5f * h, cx = r0 + 0.5f * w;
  float dy = loc[i*4+0], dx = loc[i*4+1], dh = loc[i*4+2], dw = loc[i*4+3];
  float ncy = dy * h + cy, ncx = dx * w + cx;
  float nh = __expf(dh) * h, nw = __expf(dw) * w;
  float p0 = ncy - 0.5f * nh, p1 = ncx - 0.5f * nw;
  pred_box[i*4+0] = p0; pred_box[i*4+1] = p1; pred_box[i*4+2] = nh; pred_box[i*4+3] = nw;
  nms_boxes[i*4+0] = p1; nms_boxes[i*4+1] = p0; nms_boxes[i*4+2] = p1 + nw; nms_boxes[i*4+3] = p0 + nh;
}

// ---------------- single-block bitonic sort of 8192 scores (descending) ----------------
__global__ __launch_bounds__(1024) void k_sort(const float* __restrict__ scores,
                                               int* __restrict__ order)
{
  __shared__ float s[N_ROIS];
  __shared__ int   si[N_ROIS];
  const int t = threadIdx.x;
  for (int i = t; i < N_ROIS; i += 1024) { s[i] = scores[i]; si[i] = i; }
  __syncthreads();
  for (int k = 2; k <= N_ROIS; k <<= 1) {
    for (int j = k >> 1; j > 0; j >>= 1) {
      for (int i = t; i < N_ROIS; i += 1024) {
        int ixj = i ^ j;
        if (ixj > i) {
          bool desc = ((i & k) == 0);
          bool sw = desc ? (s[i] < s[ixj]) : (s[i] > s[ixj]);
          if (sw) {
            float tf = s[i]; s[i] = s[ixj]; s[ixj] = tf;
            int   ti = si[i]; si[i] = si[ixj]; si[ixj] = ti;
          }
        }
      }
      __syncthreads();
    }
  }
  for (int i = t; i < N_ROIS; i += 1024) order[i] = si[i];
}

__global__ void k_gather_boxes(const float* __restrict__ nms_boxes, const int* __restrict__ order,
                               float* __restrict__ sboxes)
{
  int i = blockIdx.x * blockDim.x + threadIdx.x;
  if (i >= N_ROIS) return;
  int o = order[i];
  sboxes[i*4+0] = nms_boxes[o*4+0]; sboxes[i*4+1] = nms_boxes[o*4+1];
  sboxes[i*4+2] = nms_boxes[o*4+2]; sboxes[i*4+3] = nms_boxes[o*4+3];
}

// ---------------- IoU suppression bitmask: bit j of row i set iff j>i && IoU>thr ----------------
__global__ __launch_bounds__(256) void k_nms_mask(const float* __restrict__ sb,
                                                  unsigned* __restrict__ mask)
{
  const int i = blockIdx.x;
  const int t = threadIdx.x;
  float a0 = sb[i*4+0], a1 = sb[i*4+1], a2 = sb[i*4+2], a3 = sb[i*4+3];
  float aarea = (a2 - a0) * (a3 - a1);
  unsigned bits = 0u;
  const int jbase = t * 32;
  for (int b = 0; b < 32; ++b) {
    int j = jbase + b;
    if (j > i) {
      float b0 = sb[j*4+0], b1 = sb[j*4+1], b2 = sb[j*4+2], b3 = sb[j*4+3];
      float tl0 = fmaxf(a0, b0), tl1 = fmaxf(a1, b1);
      float br0 = fminf(a2, b2), br1 = fminf(a3, b3);
      float w0 = fmaxf(br0 - tl0, 0.f), w1 = fmaxf(br1 - tl1, 0.f);
      float inter = w0 * w1;
      float barea = (b2 - b0) * (b3 - b1);
      float iou = inter / (aarea + barea - inter);
      if (iou > IOU_THR) bits |= (1u << b);
    }
  }
  mask[(size_t)i * 256 + t] = bits;
}

// ---------------- sequential NMS scan (single block, keep bitset in LDS) ----------------
__global__ __launch_bounds__(256) void k_nms_scan(const unsigned* __restrict__ mask,
                                                  unsigned* __restrict__ keepbits)
{
  __shared__ unsigned keep[256];
  __shared__ int cur;
  const int t = threadIdx.x;
  keep[t] = 0xFFFFFFFFu;
  __syncthreads();
  for (int i = 0; i < N_ROIS; ++i) {
    if (t == 0) cur = (keep[i >> 5] >> (i & 31)) & 1;
    __syncthreads();
    if (cur) keep[t] &= ~mask[(size_t)i * 256 + t];
    __syncthreads();
  }
  keepbits[t] = keep[t];
}

__global__ void k_keep_scatter(const unsigned* __restrict__ keepbits, const int* __restrict__ order,
                               float* __restrict__ o_keep)
{
  int i = blockIdx.x * blockDim.x + threadIdx.x;
  if (i >= N_ROIS) return;
  unsigned k = (keepbits[i >> 5] >> (i & 31)) & 1u;
  o_keep[order[i]] = k ? 1.0f : 0.0f;
}

// ---------------- GT assignment: IoU(pred, gt), argmax, label, box2loc ----------------
__global__ __launch_bounds__(256) void k_gt_assign(
    const float* __restrict__ rois, const float* __restrict__ pred_box,
    const float* __restrict__ gt_bbox, const int* __restrict__ gt_label,
    float* __restrict__ gt_roi_loc, float* __restrict__ gt_roi_label)
{
  __shared__ float g[N_GT * 4];
  __shared__ int gl[N_GT];
  const int t = threadIdx.x;
  if (t < N_GT * 4) g[t] = gt_bbox[t];
  if (t < N_GT)     gl[t] = gt_label[t];
  __syncthreads();
  int i = blockIdx.x * blockDim.x + t;
  if (i >= N_ROIS) return;

  float p0 = pred_box[i*4+0], p1 = pred_box[i*4+1], p2 = pred_box[i*4+2], p3 = pred_box[i*4+3];
  float ay1 = p0, ax1 = p1, ay2 = p0 + p2, ax2 = p1 + p3;   // ywh -> corners
  float aarea = p2 * p3;

  float best = -1e30f; int bidx = 0;
  for (int j = 0; j < N_GT; ++j) {
    float b0 = g[j*4+0], b1 = g[j*4+1], b2 = g[j*4+2], b3 = g[j*4+3];
    float tly = fmaxf(ay1, b0),      tlx = fmaxf(ax1, b1);
    float bry = fminf(ay2, b0 + b2), brx = fminf(ax2, b1 + b3);
    float ih = fmaxf(bry - tly, 0.f), iw = fmaxf(brx - tlx, 0.f);
    float inter = ih * iw;
    float iou = inter / (aarea + b2 * b3 - inter);
    if (iou > best) { best = iou; bidx = j; }   // strict > keeps first max (argmax)
  }
  float d0 = g[bidx*4+0], d1 = g[bidx*4+1], d2 = g[bidx*4+2], d3 = g[bidx*4+3];
  int lab = (best < BG_THR) ? 0 : gl[bidx];

  float r0 = rois[i*4+0], r1 = rois[i*4+1], r2 = rois[i*4+2], r3 = rois[i*4+3];
  float hs = r3 - r1, ws_ = r2 - r0;
  float cys = r1 + 0.5f * hs, cxs = r0 + 0.5f * ws_;
  float cyd = d0 + 0.5f * d2, cxd = d1 + 0.5f * d3;
  gt_roi_loc[i*4+0] = (cyd - cys) / hs;
  gt_roi_loc[i*4+1] = (cxd - cxs) / ws_;
  gt_roi_loc[i*4+2] = __logf(d2 / hs);
  gt_roi_loc[i*4+3] = __logf(d3 / ws_);
  gt_roi_label[i] = (float)lab;
}

// ---------------------------------------------------------------------------
extern "C" void kernel_launch(void* const* d_in, const int* in_sizes, int n_in,
                              void* d_out, int out_size, void* d_ws, size_t ws_size,
                              hipStream_t stream)
{
  (void)in_sizes; (void)n_in; (void)out_size; (void)ws_size;
  const float* rpn  = (const float*)d_in[0];
  const float* rois = (const float*)d_in[1];
  const float* gtb  = (const float*)d_in[2];
  const int*   gtl  = (const int*)  d_in[3];
  const float* W1  = (const float*)d_in[4];  const float* b1  = (const float*)d_in[5];
  const float* W2  = (const float*)d_in[6];  const float* b2  = (const float*)d_in[7];
  const float* W3  = (const float*)d_in[8];  const float* b3  = (const float*)d_in[9];
  const float* Wl1 = (const float*)d_in[10]; const float* bl1 = (const float*)d_in[11];
  const float* Wl2 = (const float*)d_in[12]; const float* bl2 = (const float*)d_in[13];
  const float* Wc1 = (const float*)d_in[14]; const float* bc1 = (const float*)d_in[15];
  const float* Wc2 = (const float*)d_in[16]; const float* bc2 = (const float*)d_in[17];

  // outputs concatenated in return order: loc, cls, gt_roi_loc, gt_roi_label, pred_box, keep
  float* out    = (float*)d_out;
  float* o_loc  = out;
  float* o_cls  = o_loc  + (size_t)N_ROIS * 4;
  float* o_gloc = o_cls  + (size_t)N_ROIS * NUM_CLASSES;
  float* o_glab = o_gloc + (size_t)N_ROIS * 4;
  float* o_pbox = o_glab + (size_t)N_ROIS;
  float* o_keep = o_pbox + (size_t)N_ROIS * 4;

  // workspace carve-up
  char* p = (char*)d_ws;
  auto alloc = [&](size_t bytes) -> char* {
    char* r = p; p += (bytes + 255) & ~(size_t)255; return r;
  };
  const size_t WSZ = (size_t)IN_CH * IN_CH;      // weight elems
  const size_t ASZ = (size_t)N_ROIS * IN_CH;     // activation elems
  bf16_t* W1b  = (bf16_t*)alloc(WSZ * 2);
  bf16_t* W2b  = (bf16_t*)alloc(WSZ * 2);
  bf16_t* W3b  = (bf16_t*)alloc(WSZ * 2);
  bf16_t* Wl1b = (bf16_t*)alloc(WSZ * 2);
  bf16_t* Wc1b = (bf16_t*)alloc(WSZ * 2);
  bf16_t* X0   = (bf16_t*)alloc(ASZ * 2);
  bf16_t* X1   = (bf16_t*)alloc(ASZ * 2);
  bf16_t* X2   = (bf16_t*)alloc(ASZ * 2);
  float*  clsL = (float*)  alloc((size_t)N_ROIS * NUM_CLASSES * 4);
  float*  scor = (float*)  alloc((size_t)N_ROIS * 4);
  int*    ordr = (int*)    alloc((size_t)N_ROIS * 4);
  float*  nmsb = (float*)  alloc((size_t)N_ROIS * 4 * 4);
  float*  sbox = (float*)  alloc((size_t)N_ROIS * 4 * 4);
  unsigned* mask = (unsigned*)alloc((size_t)N_ROIS * 256 * 4);
  unsigned* kbits = (unsigned*)alloc(256 * 4);

  // --- precision conversion ---
  k_f32_to_bf16<<<1024, 256, 0, stream>>>(rpn, X0, (int)ASZ);
  k_f32_to_bf16<<<1024, 256, 0, stream>>>(W1,  W1b,  (int)WSZ);
  k_f32_to_bf16<<<1024, 256, 0, stream>>>(W2,  W2b,  (int)WSZ);
  k_f32_to_bf16<<<1024, 256, 0, stream>>>(W3,  W3b,  (int)WSZ);
  k_f32_to_bf16<<<1024, 256, 0, stream>>>(Wl1, Wl1b, (int)WSZ);
  k_f32_to_bf16<<<1024, 256, 0, stream>>>(Wc1, Wc1b, (int)WSZ);

  // --- GEMM chain (WMMA, async double-buffered) ---
  dim3 gg(IN_CH / BN, N_ROIS / BM);  // (16, 64)
  k_gemm_bf16<<<gg, 256, 0, stream>>>(X0, W1b,  b1,  nullptr, X1, N_ROIS, IN_CH, IN_CH); // h1
  k_gemm_bf16<<<gg, 256, 0, stream>>>(X1, W2b,  b2,  nullptr, X2, N_ROIS, IN_CH, IN_CH); // h2
  k_gemm_bf16<<<gg, 256, 0, stream>>>(X2, W3b,  b3,  nullptr, X0, N_ROIS, IN_CH, IN_CH); // out
  k_gemm_bf16<<<gg, 256, 0, stream>>>(X0, Wl1b, bl1, nullptr, X1, N_ROIS, IN_CH, IN_CH); // loc hidden
  k_gemm_bf16<<<gg, 256, 0, stream>>>(X0, Wc1b, bc1, nullptr, X2, N_ROIS, IN_CH, IN_CH); // cls hidden

  // --- small heads ---
  k_head_gemm<<<N_ROIS, 128, 0, stream>>>(X1, Wl2, bl2, o_loc, 4, IN_CH);
  k_head_gemm<<<N_ROIS, 128, 0, stream>>>(X2, Wc2, bc2, clsL, NUM_CLASSES, IN_CH);

  // --- softmax + scores ---
  k_softmax<<<N_ROIS, 128, 0, stream>>>(clsL, o_cls, scor);

  // --- box decode ---
  k_pred<<<N_ROIS / 256, 256, 0, stream>>>(rois, o_loc, o_pbox, nmsb);

  // --- NMS ---
  k_sort<<<1, 1024, 0, stream>>>(scor, ordr);
  k_gather_boxes<<<N_ROIS / 256, 256, 0, stream>>>(nmsb, ordr, sbox);
  k_nms_mask<<<N_ROIS, 256, 0, stream>>>(sbox, mask);
  k_nms_scan<<<1, 256, 0, stream>>>(mask, kbits);
  k_keep_scatter<<<N_ROIS / 256, 256, 0, stream>>>(kbits, ordr, o_keep);

  // --- GT assignment ---
  k_gt_assign<<<N_ROIS / 256, 256, 0, stream>>>(rois, o_pbox, gtb, gtl, o_gloc, o_glab);
}
